// positate_att_57458072486021
// MI455X (gfx1250) — matrix operand order repaired and checked by reading, not compiled, not run on hardware
//
#include <hip/hip_runtime.h>
#include <hip/hip_bf16.h>

// ---------------- problem constants ----------------
#define NB   4
#define NN   512
#define NDIM 128
#define PH   64
#define AH   256

typedef __attribute__((ext_vector_type(16))) __bf16 v16bf;
typedef __attribute__((ext_vector_type(8)))  float  v8f;

// ---------------- workspace byte offsets ----------------
// packed weight blob (B-register-image layout) lives at ws+0:
#define PACK_WA1   0u        // pW2@aW1 packed: 2ks x 16nt x 32lane x 32B = 32768
#define PACK_WA2   32768u    // perm(aW2) packed: 8ks x 8nt x 1024B     = 65536
#define PACK_WP2   98304u    // pW2 packed: 2ks x 8nt x 1024B           = 16384
#define PACK_PW1   114688u   // raw f32: pW1[3][64] + pb1[64]           = 1024
#define PACK_BYTES 115712u
#define WS_QW1     115712u   // f32 [4][512][256]  (q@aW1 + pb2@aW1 + ab1)
#define WS_KW1T    2212864u  // f32 [4][256][512]  (k@aW1 transposed)
#define WS_VT      4310016u  // f32 [4][128][512]  (v transposed)
#define WS_QKV     5358592u  // f32 [4][512][384]
#define WS_PW2AW1  8504320u  // f32 [64][256]
#define WS_B1F     8569856u  // f32 [256]

__device__ __forceinline__ unsigned short f2bf(float f) {
  unsigned u = __builtin_bit_cast(unsigned, f);
  u += 0x7FFFu + ((u >> 16) & 1u);
  return (unsigned short)(u >> 16);
}

__device__ __forceinline__ v8f wmma_bf16(v16bf a, v16bf b, v8f c) {
  return __builtin_amdgcn_wmma_f32_16x16x32_bf16(false, a, false, b, (short)0, c, false, false);
}
// A-operand: 32B contiguous at p (K k0..k0+15), row-major LDS
__device__ __forceinline__ v16bf ldsA(const unsigned char* p) {
  union { uint4 u[2]; v16bf v; } x;
  x.u[0] = *(const uint4*)p; x.u[1] = *(const uint4*)(p + 16); return x.v;
}
// B-operand: 32B contiguous per lane (pre-packed register image)
__device__ __forceinline__ v16bf ldsB(const unsigned char* p) {
  union { uint4 u[2]; v16bf v; } x;
  x.u[0] = *(const uint4*)p; x.u[1] = *(const uint4*)(p + 16); return x.v;
}

// ================= pre-pass kernels =================

// blocks 0..63: pW2@aW1 row h; block 64: bias1f = pb2@aW1 + ab1
__global__ void k_pw2aw1(const float* __restrict__ pW2, const float* __restrict__ aW1,
                         const float* __restrict__ pb2, const float* __restrict__ ab1,
                         float* __restrict__ ws_f) {
  const int n = threadIdx.x, h = blockIdx.x;
  if (h < 64) {
    float acc = 0.f;
    for (int d = 0; d < NDIM; ++d) acc = fmaf(pW2[h*NDIM+d], aW1[d*AH+n], acc);
    ws_f[WS_PW2AW1/4 + h*AH + n] = acc;
  } else {
    float acc = ab1[n];
    for (int d = 0; d < NDIM; ++d) acc = fmaf(pb2[d], aW1[d*AH+n], acc);
    ws_f[WS_B1F/4 + n] = acc;
  }
}

// pack all B operands into the per-lane WMMA register image (bf16 pairs per dword)
__global__ void k_pack(const float* __restrict__ pw2aw1f, const float* __restrict__ aW2,
                       const float* __restrict__ pW2, const float* __restrict__ pW1,
                       const float* __restrict__ pb1, unsigned* __restrict__ pk) {
  const int idx = blockIdx.x*256 + threadIdx.x;      // 28928 dwords total
  if (idx < 8192) {                                  // WA1: K=64, NT=16, src [64][256]
    int t = idx; int v = t&7, l = (t>>3)&31, nt = (t>>8)&15, ks = t>>12;
    int k = ks*32 + (l>>4)*16 + 2*v, n = nt*16 + (l&15);
    pk[idx] = (unsigned)f2bf(pw2aw1f[k*AH+n]) | ((unsigned)f2bf(pw2aw1f[(k+1)*AH+n]) << 16);
  } else if (idx < 24576) {                          // WA2: K=256, NT=8, src [256][128], k-permuted
    int t = idx - 8192; int v = t&7, l = (t>>3)&31, nt = (t>>8)&7, ks = t>>11;
    int kp0 = ks*32 + (l>>4)*16 + 2*v, n = nt*16 + (l&15);
    int p0 = kp0 & 31, s0 = (kp0 & ~31) + (p0&1)*16 + (p0>>1);
    int p1 = (kp0+1) & 31, s1 = ((kp0+1) & ~31) + (p1&1)*16 + (p1>>1);
    pk[idx] = (unsigned)f2bf(aW2[s0*NDIM+n]) | ((unsigned)f2bf(aW2[s1*NDIM+n]) << 16);
  } else if (idx < 28672) {                          // WP2: K=64, NT=8, src [64][128]
    int t = idx - 24576; int v = t&7, l = (t>>3)&31, nt = (t>>8)&7, ks = t>>11;
    int k = ks*32 + (l>>4)*16 + 2*v, n = nt*16 + (l&15);
    pk[idx] = (unsigned)f2bf(pW2[k*NDIM+n]) | ((unsigned)f2bf(pW2[(k+1)*NDIM+n]) << 16);
  } else if (idx < 28928) {                          // raw f32 pW1 + pb1
    int t = idx - 28672;
    float val = (t < 192) ? pW1[t] : pb1[t-192];
    pk[idx] = __builtin_bit_cast(unsigned, val);
  }
}

// qkv[b][i][c] = x^T @ Wqkv + bqkv
__global__ void k_qkv(const float* __restrict__ x, const float* __restrict__ Wqkv,
                      const float* __restrict__ bqkv, float* __restrict__ qkv) {
  const int idx = blockIdx.x*256 + threadIdx.x;      // 4*512*384
  const int c = idx % 384, i = (idx/384) & 511, b = idx/(384*512);
  const float* xp = x + (b*NDIM)*NN + i;
  float acc = bqkv[c];
  for (int d = 0; d < NDIM; ++d) acc = fmaf(xp[d*NN], Wqkv[d*384+c], acc);
  qkv[idx] = acc;
}

// qW1[b][i][n] = q @ aW1 + bias1f
__global__ void k_qw1(const float* __restrict__ qkv, const float* __restrict__ aW1,
                      const float* __restrict__ b1f, float* __restrict__ qW1) {
  const int idx = blockIdx.x*256 + threadIdx.x;      // 4*512*256
  const int n = idx & 255, i = (idx>>8) & 511, b = idx>>17;
  const float* qp = qkv + (b*NN+i)*384;
  float acc = b1f[n];
  for (int d = 0; d < NDIM; ++d) acc = fmaf(qp[d], aW1[d*AH+n], acc);
  qW1[idx] = acc;
}

// kW1T[b][n][j] = (k @ aW1)^T
__global__ void k_kw1t(const float* __restrict__ qkv, const float* __restrict__ aW1,
                       float* __restrict__ kW1T) {
  const int idx = blockIdx.x*256 + threadIdx.x;      // 4*256*512
  const int j = idx & 511, n = (idx>>9) & 255, b = idx>>17;
  const float* kp = qkv + (b*NN+j)*384 + NDIM;
  float acc = 0.f;
  for (int d = 0; d < NDIM; ++d) acc = fmaf(kp[d], aW1[d*AH+n], acc);
  kW1T[idx] = acc;
}

// vT[b][d][j] = v^T
__global__ void k_vt(const float* __restrict__ qkv, float* __restrict__ vT) {
  const int idx = blockIdx.x*256 + threadIdx.x;      // 4*128*512
  const int j = idx & 511, d = (idx>>9) & 127, b = idx>>16;
  vT[idx] = qkv[(b*NN+j)*384 + 256 + d];
}

// ================= main fused attention kernel =================
// LDS layout (bytes)
#define LDS_WA1 0u
#define LDS_WA2 32768u
#define LDS_WP2 98304u
#define LDS_PW1 114688u
#define LDS_P1  115712u   // 8 waves * 2048 (relu(P1) tile, 16x64 bf16)
#define LDS_SCH 132096u   // 8 waves * 1024 (S chunk, 16x32 bf16)
#define LDS_RED 115712u   // alias over P1+SCH: 8*32*8*3 f32 = 24576
#define LDS_TOT 140288u

__launch_bounds__(256)
__global__ void k_main(const float* __restrict__ pos, const float* __restrict__ pb2,
                       const unsigned char* __restrict__ ws, float* __restrict__ out) {
  __shared__ unsigned char lds[LDS_TOT];
  const int tid = threadIdx.x;
  const int wave = tid >> 5, lane = tid & 31, half = lane >> 4, lo = lane & 15;
  const int blk = blockIdx.x, b = blk >> 9, i = blk & 511;

  // cooperative copy of packed weights -> LDS
  for (unsigned ofs = (unsigned)tid * 16u; ofs < PACK_BYTES; ofs += 4096u)
    *(uint4*)(lds + ofs) = *(const uint4*)(ws + ofs);
  __syncthreads();

  const float* qW1  = (const float*)(ws + WS_QW1);
  const float* kW1T = (const float*)(ws + WS_KW1T);
  const float* vT   = (const float*)(ws + WS_VT);
  const float* w1   = (const float*)(lds + LDS_PW1);

  float qp[16];
  #pragma unroll
  for (int t = 0; t < 16; ++t) qp[t] = qW1[((b<<9)+i)*AH + t*16 + lo];

  const float pix = pos[((b<<9)+i)*3+0];
  const float piy = pos[((b<<9)+i)*3+1];
  const float piz = pos[((b<<9)+i)*3+2];

  float m_s[8], l_s[8], a_s[8];
  #pragma unroll
  for (int t = 0; t < 8; ++t) { m_s[t] = -3.0e38f; l_s[t] = 0.f; a_s[t] = 0.f; }

  for (int jt = wave; jt < 32; jt += 8) {
    const int j0 = jt << 4;
    // ---- P1 = relu(rel_pos @ pW1 + pb1): lane -> row j=lo, cols half*32..+31
    {
      const int jg = j0 + lo;
      const float rx = pix - pos[((b<<9)+jg)*3+0];
      const float ry = piy - pos[((b<<9)+jg)*3+1];
      const float rz = piz - pos[((b<<9)+jg)*3+2];
      __attribute__((aligned(16))) unsigned pkw[16];
      #pragma unroll
      for (int hh = 0; hh < 32; ++hh) {
        const int h = half*32 + hh;
        float v = fmaf(rx, w1[h], fmaf(ry, w1[64+h], fmaf(rz, w1[128+h], w1[192+h])));
        v = fmaxf(v, 0.f);
        const unsigned bv = f2bf(v);
        if (hh & 1) pkw[hh>>1] |= bv << 16; else pkw[hh>>1] = bv;
      }
      uint4* dst = (uint4*)(lds + LDS_P1 + (wave<<11) + lo*128 + half*64);
      #pragma unroll
      for (int q = 0; q < 4; ++q) dst[q] = ((uint4*)pkw)[q];
    }
    // ---- A operand (relu(P1)) for both K-steps (reused by posGEMM and GEMM1')
    v16bf aA[2];
    #pragma unroll
    for (int ks = 0; ks < 2; ++ks)
      aA[ks] = ldsA(lds + LDS_P1 + (wave<<11) + lo*128 + ks*64 + half*32);

    // ---- posGEMM: rel_emb; vv = rel_emb + v_j  (kept in registers)
    v8f vv[8];
    #pragma unroll
    for (int t = 0; t < 8; ++t) {
      v8f acc = {};
      #pragma unroll
      for (int ks = 0; ks < 2; ++ks)
        acc = wmma_bf16(aA[ks], ldsB(lds + LDS_WP2 + ((((ks<<3)+t)<<5) + lane)*32u), acc);
      const float* vp = vT + ((b<<7) + t*16 + lo)*NN + j0 + half*8;
      const float4 va = *(const float4*)vp;
      const float4 vb = *(const float4*)(vp + 4);
      acc[0]+=va.x; acc[1]+=va.y; acc[2]+=va.z; acc[3]+=va.w;
      acc[4]+=vb.x; acc[5]+=vb.y; acc[6]+=vb.z; acc[7]+=vb.w;
      vv[t] = acc;
    }

    // ---- GEMM1' (K=64,N=256) chunk-by-chunk, fused into GEMM2 (K=256,N=128)
    v8f s2[8];
    #pragma unroll
    for (int t = 0; t < 8; ++t) s2[t] = (v8f){};
    for (int c = 0; c < 8; ++c) {
      v8f a0 = {}, a1 = {};
      #pragma unroll
      for (int ks = 0; ks < 2; ++ks) {
        a0 = wmma_bf16(aA[ks], ldsB(lds + LDS_WA1 + ((((ks<<4)+2*c  )<<5) + lane)*32u), a0);
        a1 = wmma_bf16(aA[ks], ldsB(lds + LDS_WA1 + ((((ks<<4)+2*c+1)<<5) + lane)*32u), a1);
      }
      const float* kp0 = kW1T + ((b<<8) + (2*c)*16 + lo)*NN + j0 + half*8;
      const float* kp1 = kp0 + 16*NN;
      unsigned char* sb = lds + LDS_SCH + (wave<<10) + (half*8)*64 + lo*4;
      #pragma unroll
      for (int r = 0; r < 8; ++r) {
        const float f0 = fmaxf(a0[r] + qp[2*c]   - kp0[r], 0.f);
        const float f1 = fmaxf(a1[r] + qp[2*c+1] - kp1[r], 0.f);
        *(unsigned*)(sb + r*64) = (unsigned)f2bf(f0) | ((unsigned)f2bf(f1) << 16);
      }
      // read S chunk back as A operand (k-permutation matches packed aW2)
      const v16bf aS = ldsA(lds + LDS_SCH + (wave<<10) + lo*64 + half*32);
      #pragma unroll
      for (int t = 0; t < 8; ++t)
        s2[t] = wmma_bf16(aS, ldsB(lds + LDS_WA2 + ((((c<<3)+t)<<5) + lane)*32u), s2[t]);
    }

    // ---- online softmax update (per lane: 8 j-rows for fixed d = t*16+lo)
    #pragma unroll
    for (int t = 0; t < 8; ++t) {
      float mx = m_s[t];
      #pragma unroll
      for (int r = 0; r < 8; ++r) mx = fmaxf(mx, s2[t][r]);
      const float sc = __expf(m_s[t] - mx);
      float ls = l_s[t]*sc, as = a_s[t]*sc;
      #pragma unroll
      for (int r = 0; r < 8; ++r) {
        const float e = __expf(s2[t][r] - mx);
        ls += e; as += e * vv[t][r];
      }
      m_s[t] = mx; l_s[t] = ls; a_s[t] = as;
    }
  }

  // ---- cross-wave combine of (m,l,acc) partials
  __syncthreads();
  float* red = (float*)(lds + LDS_RED);
  #pragma unroll
  for (int t = 0; t < 8; ++t) {
    const int base = (((wave<<5)+lane)*8 + t)*3;
    red[base+0] = m_s[t]; red[base+1] = l_s[t]; red[base+2] = a_s[t];
  }
  __syncthreads();
  if (tid < 128) {
    const int d = tid;
    float M = -3.0e38f;
    #pragma unroll
    for (int w = 0; w < 8; ++w)
      #pragma unroll
      for (int h = 0; h < 2; ++h)
        M = fmaxf(M, red[(((w<<5) + (d&15) + (h<<4))*8 + (d>>4))*3]);
    float L = 0.f, A = 0.f;
    #pragma unroll
    for (int w = 0; w < 8; ++w)
      #pragma unroll
      for (int h = 0; h < 2; ++h) {
        const int p = (((w<<5) + (d&15) + (h<<4))*8 + (d>>4))*3;
        const float sc = __expf(red[p] - M);
        L += red[p+1]*sc; A += red[p+2]*sc;
      }
    out[((b<<7)+d)*NN + i] = A / L + pb2[d];   // +pb2 (softmax-weight sum == 1)
  }
}

// ================= launch =================
extern "C" void kernel_launch(void* const* d_in, const int* in_sizes, int n_in,
                              void* d_out, int out_size, void* d_ws, size_t ws_size,
                              hipStream_t stream) {
  (void)in_sizes; (void)n_in; (void)out_size; (void)ws_size;
  const float* x    = (const float*)d_in[0];
  const float* pos  = (const float*)d_in[1];
  const float* Wqkv = (const float*)d_in[2];
  const float* bqkv = (const float*)d_in[3];
  const float* pW1  = (const float*)d_in[4];
  const float* pb1  = (const float*)d_in[5];
  const float* pW2  = (const float*)d_in[6];
  const float* pb2  = (const float*)d_in[7];
  const float* aW1  = (const float*)d_in[8];
  const float* ab1  = (const float*)d_in[9];
  const float* aW2  = (const float*)d_in[10];
  // d_in[11] == ab2: constant over j -> softmax-invariant -> unused.
  unsigned char* ws = (unsigned char*)d_ws;

  k_pw2aw1<<<65,   256, 0, stream>>>(pW2, aW1, pb2, ab1, (float*)ws);
  k_pack  <<<113,  256, 0, stream>>>((const float*)(ws + WS_PW2AW1), aW2, pW2, pW1, pb1,
                                     (unsigned*)ws);
  k_qkv   <<<3072, 256, 0, stream>>>(x, Wqkv, bqkv, (float*)(ws + WS_QKV));
  k_qw1   <<<2048, 256, 0, stream>>>((const float*)(ws + WS_QKV), aW1,
                                     (const float*)(ws + WS_B1F), (float*)(ws + WS_QW1));
  k_kw1t  <<<2048, 256, 0, stream>>>((const float*)(ws + WS_QKV), aW1, (float*)(ws + WS_KW1T));
  k_vt    <<<1024, 256, 0, stream>>>((const float*)(ws + WS_QKV), (float*)(ws + WS_VT));
  k_main  <<<2048, 256, 0, stream>>>(pos, pb2, ws, (float*)d_out);
}
